// RNN_29618094473480
// MI455X (gfx1250) — compile-verified, matrix operand order checked by reading, not run on hardware
//
#include <hip/hip_runtime.h>

typedef __attribute__((ext_vector_type(16))) _Float16 v16h;
typedef __attribute__((ext_vector_type(2)))  _Float16 h2;
typedef __attribute__((ext_vector_type(2)))  __fp16   fp16x2;
typedef __attribute__((ext_vector_type(8)))  float    v8f;
typedef __attribute__((ext_vector_type(4)))  float    v4f;

#define TT 512
#define HH 64

__device__ __forceinline__ float ftanh(float x) {
#if __has_builtin(__builtin_amdgcn_tanhf)
  return __builtin_amdgcn_tanhf(x);          // v_tanh_f32 (new gfx1250 trans op)
#else
  // 2/(1+exp(-2x)) - 1 : safe at both infinities (no inf/inf).
  float e = __builtin_amdgcn_exp2f(x * -2.88539008177792681472f);
  return __builtin_fmaf(2.0f, __builtin_amdgcn_rcpf(1.0f + e), -1.0f);
#endif
}
__device__ __forceinline__ float fsig(float x) {
  // sigmoid(x) = 0.5*tanh(x/2) + 0.5  -> 1 trans op instead of exp+rcp
  return __builtin_fmaf(0.5f, ftanh(0.5f * x), 0.5f);
}

__device__ __forceinline__ h2 pack2(float a, float b) {
#if __has_builtin(__builtin_amdgcn_cvt_pkrtz)
  fp16x2 t = __builtin_amdgcn_cvt_pkrtz(a, b);   // v_cvt_pk_rtz_f16_f32
  return __builtin_bit_cast(h2, t);
#else
  h2 r; r[0] = (_Float16)a; r[1] = (_Float16)b; return r;
#endif
}

__device__ __forceinline__ v8f wmma_f16(v16h a, v16h b, v8f c) {
  // D = A(16x32 f16) * B(32x16 f16) + C(16x16 f32)
  return __builtin_amdgcn_wmma_f32_16x16x32_f16(false, a, false, b, (short)0, c,
                                                false, false);
}

// Block = 128 threads = 4 waves; block owns 16 batch rows for all T steps.
// Wave w computes gate tiles m in {w, 4+w, 8+w, 12+w}  (i/f/g/o rows of
// hidden block w) and owns c-tile w. h is exchanged between waves through a
// double-buffered LDS region stored directly in WMMA B-fragment layout.
__global__ __launch_bounds__(128) void lstm_wmma_kernel(
    const float* __restrict__ x,      // [B, T] (I == 1)
    const float* __restrict__ W_ih,   // [256]
    const float* __restrict__ W_hh,   // [256, 64]
    const float* __restrict__ b_ih,   // [256]
    const float* __restrict__ b_hh,   // [256]
    const float* __restrict__ W_out,  // [3, 64]
    const float* __restrict__ b_out,  // [3]
    float* __restrict__ out)          // [B, 3]
{
  // 2 x (2 fragments x 32 lanes x 16 halves) = 2 x 2048 B, double buffered.
  __shared__ __align__(16) char  hbuf[2][2048];
  __shared__ float hlast[16 * HH];   // f32 h_T for the output head

  const int tid    = threadIdx.x;
  const int lane   = tid & 31;
  const int w      = tid >> 5;       // wave id 0..3
  const int nlo    = lane & 15;      // batch column within tile (WMMA N / M-of-A)
  const int s      = lane >> 4;      // half-wave selector
  const int batch0 = blockIdx.x * 16;
  const int my_b   = batch0 + nlo;

  // ---- cache this wave's W_hh A-fragments + x-projection constants in regs ----
  v16h A[4][2];                      // [gate i/f/g/o][k-tile]
  float wib[4][8], bia[4][8];
  #pragma unroll
  for (int gi = 0; gi < 4; ++gi) {
    const int m = 4 * gi + w;        // gate tile index (rows 16m..16m+15)
    #pragma unroll
    for (int kt = 0; kt < 2; ++kt) {
      // A layout (16-bit 16x32): lane -> M = lane&15; halves p<8 -> K = 8s+p,
      // p>=8 -> K = 16+8s+(p-8); plus 32*kt. Rows of W_hh are K-contiguous.
      const float* p0 = W_hh + (m * 16 + nlo) * HH + kt * 32 + s * 8;
      union { v16h h; _Float16 e[16]; } u;
      #pragma unroll
      for (int i = 0; i < 8; ++i) {
        u.e[i]     = (_Float16)p0[i];
        u.e[8 + i] = (_Float16)p0[16 + i];
      }
      A[gi][kt] = u.h;
    }
    #pragma unroll
    for (int r = 0; r < 8; ++r) {
      const int g = m * 16 + s * 8 + r;   // D layout: vgpr r -> row r + 8s
      wib[gi][r] = W_ih[g];
      bia[gi][r] = b_ih[g] + b_hh[g];
    }
  }

  // ---- h0 = 0: each wave zero-fills its quarter of buffer 0 ----
  {
    v4f z = {0.f, 0.f, 0.f, 0.f};
    *(v4f*)(hbuf[0] + (w >> 1) * 1024 + (w & 1) * 512 + lane * 16) = z;
  }
  __syncthreads();

  const float* xb = x + my_b * TT;
  float xv = xb[0];
  v8f c = {};                        // c-tile w, D layout, f32

  for (int t = 0; t < TT; ++t) {
    // software-pipeline next x (clamped index: compiler peels the tail,
    // never speculates past xb[TT-1])
    float xn = xb[(t + 1 < TT) ? (t + 1) : (TT - 1)];

    // read h_t as two f16 B fragments (produced by all 4 waves last step)
    const char* rb = hbuf[t & 1];
    union { v16h h; v4f q[2]; } hb0, hb1;
    hb0.q[0] = *(const v4f*)(rb + lane * 16);
    hb0.q[1] = *(const v4f*)(rb + 512  + lane * 16);
    hb1.q[0] = *(const v4f*)(rb + 1024 + lane * 16);
    hb1.q[1] = *(const v4f*)(rb + 1536 + lane * 16);

    // C init = x * W_ih + (b_ih + b_hh), then gates += W_hh x h  (8 WMMAs)
    v8f gI, gF, gG, gO;
    #pragma unroll
    for (int r = 0; r < 8; ++r) {
      gI[r] = __builtin_fmaf(xv, wib[0][r], bia[0][r]);
      gF[r] = __builtin_fmaf(xv, wib[1][r], bia[1][r]);
      gG[r] = __builtin_fmaf(xv, wib[2][r], bia[2][r]);
      gO[r] = __builtin_fmaf(xv, wib[3][r], bia[3][r]);
    }
    gI = wmma_f16(A[0][0], hb0.h, gI); gI = wmma_f16(A[0][1], hb1.h, gI);
    gF = wmma_f16(A[1][0], hb0.h, gF); gF = wmma_f16(A[1][1], hb1.h, gF);
    gG = wmma_f16(A[2][0], hb0.h, gG); gG = wmma_f16(A[2][1], hb1.h, gG);
    gO = wmma_f16(A[3][0], hb0.h, gO); gO = wmma_f16(A[3][1], hb1.h, gO);

    // LSTM cell elementwise; native v_tanh_f32 keeps the serial chain short.
    float hv[8];
    #pragma unroll
    for (int r = 0; r < 8; ++r) {
      float iv = fsig(gI[r]);
      float fv = fsig(gF[r]);
      float gv = ftanh(gG[r]);
      float ov = fsig(gO[r]);
      float cn = __builtin_fmaf(fv, c[r], iv * gv);
      c[r] = cn;
      hv[r] = ov * ftanh(cn);
      if (t == TT - 1) hlast[nlo * HH + w * 16 + s * 8 + r] = hv[r];
    }
    // D-tile M-striping == B-fragment K-striping: pack in-lane, 4 pk-cvts.
    union { v4f q; h2 p[4]; } hw;
    #pragma unroll
    for (int i = 0; i < 4; ++i) hw.p[i] = pack2(hv[2 * i], hv[2 * i + 1]);

    // publish this wave's 8 halves of h_{t+1} into the other buffer
    *(v4f*)(hbuf[(t + 1) & 1] + (w >> 1) * 1024 + (w & 1) * 512 + lane * 16) = hw.q;

    xv = xn;
    __syncthreads();   // one split-barrier per timestep
  }

  // ---- output head: out[b, :3] = h_T @ W_out.T + b_out ----
  if (tid < 48) {
    const int bl = tid / 3, o = tid % 3;
    const float* hp = hlast + bl * HH;
    const float* wp = W_out + o * HH;
    float acc = b_out[o];
    #pragma unroll 8
    for (int k = 0; k < HH; ++k) acc = __builtin_fmaf(hp[k], wp[k], acc);
    out[(batch0 + bl) * 3 + o] = acc;
  }
}

extern "C" void kernel_launch(void* const* d_in, const int* in_sizes, int n_in,
                              void* d_out, int out_size, void* d_ws, size_t ws_size,
                              hipStream_t stream) {
  (void)in_sizes; (void)n_in; (void)out_size; (void)d_ws; (void)ws_size;
  const float* x     = (const float*)d_in[0];
  const float* W_ih  = (const float*)d_in[1];
  const float* W_hh  = (const float*)d_in[2];
  const float* b_ih  = (const float*)d_in[3];
  const float* b_hh  = (const float*)d_in[4];
  const float* W_out = (const float*)d_in[5];
  const float* b_out = (const float*)d_in[6];
  float* out = (float*)d_out;

  dim3 grid(4096 / 16);   // 256 blocks x 4 waves = 1024 waves
  dim3 block(128);
  lstm_wmma_kernel<<<grid, block, 0, stream>>>(x, W_ih, W_hh, b_ih, b_hh,
                                               W_out, b_out, out);
}
